// decoder_85031762526824
// MI455X (gfx1250) — matrix operand order repaired and checked by reading, not compiled, not run on hardware
//
#include <hip/hip_runtime.h>
#include <hip/hip_bf16.h>
#include <cstddef>

// Problem constants (from reference): B=512, T=128, E=256, D=H=256, L=3
#define BATCH 512
#define T1    127
#define EDIM  256
#define HDIM  256

typedef __attribute__((ext_vector_type(8)))  float   v8f;
typedef __attribute__((ext_vector_type(16))) __bf16  v16bf;
typedef __attribute__((ext_vector_type(8)))  __bf16  v8bf;

// ---------- helpers ----------
__device__ __forceinline__ unsigned short f2bfu(float f) {
  unsigned int u = __builtin_bit_cast(unsigned int, f);
  u += 0x7FFFu + ((u >> 16) & 1u);            // round-to-nearest-even
  return (unsigned short)(u >> 16);
}
__device__ __forceinline__ __bf16 us2bf(unsigned short h) {
  __bf16 b; __builtin_memcpy(&b, &h, 2); return b;
}
__device__ __forceinline__ float sigmoidf_(float x) { return 1.0f / (1.0f + expf(-x)); }

// ---------- setup kernels ----------
// dst[r*cols + c] = bf16(src[r*src_stride + col0 + c])
__global__ void pack_slice(const float* __restrict__ src, unsigned short* __restrict__ dst,
                           int rows, int src_stride, int col0, int cols) {
  int idx = blockIdx.x * 256 + threadIdx.x;
  if (idx >= rows * cols) return;
  int r = idx / cols, c = idx % cols;
  dst[idx] = f2bfu(src[(size_t)r * src_stride + col0 + c]);
}

// out (N x (Ka+Kb)) = bf16([a | b]) with a (N x Ka), b (N x Kb)
__global__ void pack2(const float* __restrict__ a, const float* __restrict__ b,
                      unsigned short* __restrict__ out, int N_, int Ka, int Kb) {
  int idx = blockIdx.x * 256 + threadIdx.x;
  int Kt = Ka + Kb;
  if (idx >= N_ * Kt) return;
  int n = idx / Kt, k = idx % Kt;
  float v = (k < Ka) ? a[(size_t)n * Ka + k] : b[(size_t)n * Kb + (k - Ka)];
  out[idx] = f2bfu(v);
}

__global__ void zero_u32(unsigned int* __restrict__ p, int n) {
  int i = blockIdx.x * 256 + threadIdx.x;
  if (i < n) p[i] = 0u;
}

// ---------- bf16 WMMA GEMM: C(MxN,f32) = A(MxK) @ B(NxK)^T (+bias[N]) ----------
// One wave per 16x64 output strip (4 WMMA tiles share one A fragment),
// 4 waves per block -> block covers 64x64 of C.
// A-fragment (16x32 bf16, ISA layout): lane L (half=L/16, m=L%16) holds
//   k in [kb+8*half, +8) and [kb+16+8*half, +8)  -> two 16B contiguous loads.
// B-fragment (32x16): lane L (n=L%16) holds k in [kb+8*half, +8) and [kb+16+8*half, +8)
//   -> two 16B contiguous loads from the N x K row-major weight matrix.
// D: element v of lane L = C[tm + v + 8*half][tn + L%16].
template <bool A_IS_F32>
__global__ __launch_bounds__(128) void gemm_wmma_bf16(
    float* __restrict__ C, const void* __restrict__ Av,
    const unsigned short* __restrict__ Bw, const float* __restrict__ bias,
    int M, int N, int K)
{
  const int lane = threadIdx.x & 31;
  const int wave = threadIdx.x >> 5;
  const int tm = (blockIdx.y * 4 + wave) * 16;  // 16-row strip
  const int tn = blockIdx.x * 64;               // 64-col strip (4 tiles)
  if (tm >= M) return;

  const int half = lane >> 4;      // 0 or 1
  const int l16  = lane & 15;
  const int arow = tm + l16;       // A row this lane gathers

  v8f acc[4];
  #pragma unroll
  for (int j = 0; j < 4; ++j) {
    const float bv = bias ? bias[tn + 16 * j + l16] : 0.0f;
    #pragma unroll
    for (int i = 0; i < 8; ++i) acc[j][i] = bv;
  }

  const size_t brow0 = (size_t)(tn + l16) * K;  // lane's B row in tile j is brow0 + j*16*K

  for (int kb = 0; kb < K; kb += 32) {
    // ---- A fragment (shared by all 4 tiles) ----
    v16bf afrag;
    if (A_IS_F32) {
      const float* A  = (const float*)Av;
      const float* ap = A + (size_t)arow * K + kb + 8 * half;
      v8f lo = *(const v8f*)(ap);
      v8f hi = *(const v8f*)(ap + 16);
      #pragma unroll
      for (int i = 0; i < 8; ++i) {
        afrag[i]     = us2bf(f2bfu(lo[i]));
        afrag[i + 8] = us2bf(f2bfu(hi[i]));
      }
    } else {
      const unsigned short* A  = (const unsigned short*)Av;
      const unsigned short* ap = A + (size_t)arow * K + kb + 8 * half;
      v8bf lo = *(const v8bf*)(ap);
      v8bf hi = *(const v8bf*)(ap + 16);
      #pragma unroll
      for (int i = 0; i < 8; ++i) { afrag[i] = lo[i]; afrag[i + 8] = hi[i]; }
    }

    // ---- 4 B fragments, issue loads first, then 4 independent WMMAs ----
    v16bf bfrag[4];
    #pragma unroll
    for (int j = 0; j < 4; ++j)
      bfrag[j] = __builtin_shufflevector(*(const v8bf*)(Bw + brow0 + (size_t)j * 16 * K + kb + 8 * half), *(const v8bf*)(Bw + brow0 + (size_t)j * 16 * K + kb + 16 + 8 * half), 0, 1, 2, 3, 4, 5, 6, 7, 8, 9, 10, 11, 12, 13, 14, 15);

    #pragma unroll
    for (int j = 0; j < 4; ++j)
      acc[j] = __builtin_amdgcn_wmma_f32_16x16x32_bf16(
          /*neg_a=*/false, afrag, /*neg_b=*/false, bfrag[j],
          /*c_mod=*/(short)0, acc[j], /*reuse_a=*/false, /*reuse_b=*/false);
  }

  #pragma unroll
  for (int j = 0; j < 4; ++j)
    #pragma unroll
    for (int v = 0; v < 8; ++v)
      C[(size_t)(tm + v + 8 * half) * N + tn + 16 * j + l16] = acc[j][v];
}

// ---------- fused attention: scores -> softmax -> context -> y_tilde ----------
// one block per batch row; 256 threads = 8 waves (wave32)
__global__ __launch_bounds__(256) void attn_kernel(
    const float* __restrict__ qproj,    // B x E   (q @ W1q^T, no bias)
    const float* __restrict__ encproj,  // B x T1 x E (includes attn_b1)
    const float* __restrict__ enc,      // B x T1 x E (input_encoded)
    const float* __restrict__ w2,       // E
    const float* __restrict__ b2,       // 1
    const float* __restrict__ fcw,      // E+1
    const float* __restrict__ fcb,      // 1
    const float* __restrict__ yhist,    // B x T1
    int t,
    float* __restrict__ context,        // B x E
    float* __restrict__ ytilde)         // B
{
  const int b    = blockIdx.x;
  const int tid  = threadIdx.x;
  const int wave = tid >> 5;
  const int lane = tid & 31;

  __shared__ float s_q[EDIM];
  __shared__ float s_w2[EDIM];
  __shared__ float s_sc[T1];
  __shared__ float s_red[256];

  s_q[tid]  = qproj[(size_t)b * EDIM + tid];
  s_w2[tid] = w2[tid];
  __syncthreads();

  // scores[tt] = sum_f tanh(qp[f] + encproj[b,tt,f]) * w2[f] + b2
  for (int tt = wave; tt < T1; tt += 8) {
    const float* ep = encproj + ((size_t)b * T1 + tt) * EDIM;
    float s = 0.0f;
    #pragma unroll
    for (int i = 0; i < 8; ++i) {
      int f = lane + 32 * i;
      s += tanhf(s_q[f] + ep[f]) * s_w2[f];
    }
    #pragma unroll
    for (int m = 16; m > 0; m >>= 1) s += __shfl_xor(s, m, 32);
    if (lane == 0) s_sc[tt] = s + b2[0];
  }
  __syncthreads();

  // softmax over T1 (wave 0)
  if (wave == 0) {
    float mx = -3.402823e38f;
    for (int tt = lane; tt < T1; tt += 32) mx = fmaxf(mx, s_sc[tt]);
    #pragma unroll
    for (int m = 16; m > 0; m >>= 1) mx = fmaxf(mx, __shfl_xor(mx, m, 32));
    float sm = 0.0f;
    for (int tt = lane; tt < T1; tt += 32) {
      float e = expf(s_sc[tt] - mx);
      s_sc[tt] = e; sm += e;
    }
    #pragma unroll
    for (int m = 16; m > 0; m >>= 1) sm += __shfl_xor(sm, m, 32);
    float inv = 1.0f / sm;
    for (int tt = lane; tt < T1; tt += 32) s_sc[tt] *= inv;
  }
  __syncthreads();

  // context[f] = sum_tt alpha[tt] * enc[b,tt,f]   (coalesced across threads)
  float c = 0.0f;
  const float* eb = enc + (size_t)b * T1 * EDIM;
  for (int tt = 0; tt < T1; ++tt) c += s_sc[tt] * eb[(size_t)tt * EDIM + tid];
  context[(size_t)b * EDIM + tid] = c;

  // y_tilde = context . fcw[0:E] + y_t * fcw[E] + fcb
  s_red[tid] = c * fcw[tid];
  __syncthreads();
  #pragma unroll
  for (int st = 128; st >= 1; st >>= 1) {
    if (tid < st) s_red[tid] += s_red[tid + st];
    __syncthreads();
  }
  if (tid == 0)
    ytilde[b] = s_red[0] + yhist[(size_t)b * T1 + t] * fcw[EDIM] + fcb[0];
}

// ---------- LSTM pointwise, layer 0 ----------
__global__ __launch_bounds__(256) void lstm_pw0(
    const float* __restrict__ g,       // B x 4H (h0 @ whh0^T)
    const float* __restrict__ ytilde,  // B
    const float* __restrict__ wih0,    // 4H
    const float* __restrict__ bih,     // 4H
    const float* __restrict__ bhh,     // 4H
    float* __restrict__ c0,            // B x H
    float* __restrict__ h0f,           // B x H (f32, for final projection)
    unsigned short* __restrict__ h0bf, // B x H (next-step L0 GEMM operand)
    unsigned short* __restrict__ qbf,  // B x 2H ([h0, c0] for next-step qproj)
    unsigned short* __restrict__ xh1)  // B x 2H (write x slot [0,H))
{
  int idx = blockIdx.x * 256 + threadIdx.x;
  int b = idx / HDIM, j = idx % HDIM;
  float yt = ytilde[b];
  const float* gb = g + (size_t)b * 4 * HDIM;
  float gi = gb[j]            + yt * wih0[j]            + bih[j]            + bhh[j];
  float gf = gb[HDIM + j]     + yt * wih0[HDIM + j]     + bih[HDIM + j]     + bhh[HDIM + j];
  float gg = gb[2 * HDIM + j] + yt * wih0[2 * HDIM + j] + bih[2 * HDIM + j] + bhh[2 * HDIM + j];
  float go = gb[3 * HDIM + j] + yt * wih0[3 * HDIM + j] + bih[3 * HDIM + j] + bhh[3 * HDIM + j];
  float cn = sigmoidf_(gf) * c0[idx] + sigmoidf_(gi) * tanhf(gg);
  float hn = sigmoidf_(go) * tanhf(cn);
  c0[idx] = cn;
  h0f[idx] = hn;
  unsigned short hb = f2bfu(hn);
  h0bf[idx] = hb;
  qbf[(size_t)b * 2 * HDIM + j]        = hb;
  qbf[(size_t)b * 2 * HDIM + HDIM + j] = f2bfu(cn);
  xh1[(size_t)b * 2 * HDIM + j]        = hb;
}

// ---------- LSTM pointwise, layers 1/2 ----------
__global__ __launch_bounds__(256) void lstm_pw12(
    const float* __restrict__ g,             // B x 4H (xh @ [Wih|Whh]^T)
    const float* __restrict__ bih,           // 4H
    const float* __restrict__ bhh,           // 4H
    float* __restrict__ cl,                  // B x H
    unsigned short* __restrict__ xh_self,    // B x 2H (write h slot [H,2H))
    unsigned short* __restrict__ xh_next)    // B x 2H (write x slot [0,H)) or null
{
  int idx = blockIdx.x * 256 + threadIdx.x;
  int b = idx / HDIM, j = idx % HDIM;
  const float* gb = g + (size_t)b * 4 * HDIM;
  float gi = gb[j]            + bih[j]            + bhh[j];
  float gf = gb[HDIM + j]     + bih[HDIM + j]     + bhh[HDIM + j];
  float gg = gb[2 * HDIM + j] + bih[2 * HDIM + j] + bhh[2 * HDIM + j];
  float go = gb[3 * HDIM + j] + bih[3 * HDIM + j] + bhh[3 * HDIM + j];
  float cn = sigmoidf_(gf) * cl[idx] + sigmoidf_(gi) * tanhf(gg);
  float hn = sigmoidf_(go) * tanhf(cn);
  cl[idx] = cn;
  unsigned short hb = f2bfu(hn);
  xh_self[(size_t)b * 2 * HDIM + HDIM + j] = hb;
  if (xh_next) xh_next[(size_t)b * 2 * HDIM + j] = hb;
}

// ---------- final projection: y = [h0, context] @ fcf_w^T + fcf_b ----------
__global__ __launch_bounds__(256) void final_kernel(
    const float* __restrict__ h0f, const float* __restrict__ ctx,
    const float* __restrict__ fcfw, const float* __restrict__ fcfb,
    float* __restrict__ out)
{
  int b = blockIdx.x, tid = threadIdx.x;
  __shared__ float red[256];
  red[tid] = h0f[(size_t)b * HDIM + tid] * fcfw[tid]
           + ctx[(size_t)b * EDIM + tid] * fcfw[HDIM + tid];
  __syncthreads();
  #pragma unroll
  for (int st = 128; st >= 1; st >>= 1) {
    if (tid < st) red[tid] += red[tid + st];
    __syncthreads();
  }
  if (tid == 0) out[b] = red[0] + fcfb[0];
}

// ---------- host orchestration ----------
extern "C" void kernel_launch(void* const* d_in, const int* in_sizes, int n_in,
                              void* d_out, int out_size, void* d_ws, size_t ws_size,
                              hipStream_t stream) {
  (void)in_sizes; (void)n_in; (void)out_size; (void)ws_size;

  const float* input_encoded = (const float*)d_in[0];
  const float* y_history     = (const float*)d_in[1];
  const float* attn_w1       = (const float*)d_in[2];
  const float* attn_b1       = (const float*)d_in[3];
  const float* attn_w2       = (const float*)d_in[4];
  const float* attn_b2       = (const float*)d_in[5];
  const float* lstm_wih0     = (const float*)d_in[6];
  const float* lstm_wih_rest = (const float*)d_in[7];
  const float* lstm_whh      = (const float*)d_in[8];
  const float* lstm_bih      = (const float*)d_in[9];
  const float* lstm_bhh      = (const float*)d_in[10];
  const float* fc_w          = (const float*)d_in[11];
  const float* fc_b          = (const float*)d_in[12];
  const float* fcf_w         = (const float*)d_in[13];
  const float* fcf_b         = (const float*)d_in[14];

  // workspace carve-up (256B aligned), ~76 MB total
  char* ws = (char*)d_ws;
  size_t off = 0;
  auto carve = [&](size_t bytes) -> char* {
    char* p = ws + off;
    off = (off + bytes + 255) & ~(size_t)255;
    return p;
  };
  float*          encproj = (float*)         carve((size_t)BATCH * T1 * EDIM * 4);
  unsigned short* w1e     = (unsigned short*)carve((size_t)EDIM * EDIM * 2);
  unsigned short* w1q     = (unsigned short*)carve((size_t)EDIM * 2 * HDIM * 2);
  unsigned short* whh0    = (unsigned short*)carve((size_t)4 * HDIM * HDIM * 2);
  unsigned short* W1      = (unsigned short*)carve((size_t)4 * HDIM * 2 * HDIM * 2);
  unsigned short* W2      = (unsigned short*)carve((size_t)4 * HDIM * 2 * HDIM * 2);
  unsigned short* qbf     = (unsigned short*)carve((size_t)BATCH * 2 * HDIM * 2);
  unsigned short* h0bf    = (unsigned short*)carve((size_t)BATCH * HDIM * 2);
  unsigned short* xh1     = (unsigned short*)carve((size_t)BATCH * 2 * HDIM * 2);
  unsigned short* xh2     = (unsigned short*)carve((size_t)BATCH * 2 * HDIM * 2);
  float*          qproj   = (float*)         carve((size_t)BATCH * EDIM * 4);
  float*          ctx     = (float*)         carve((size_t)BATCH * EDIM * 4);
  float*          ytl     = (float*)         carve((size_t)BATCH * 4);
  float*          g       = (float*)         carve((size_t)BATCH * 4 * HDIM * 4);
  float*          c0      = (float*)         carve((size_t)BATCH * HDIM * 4);
  float*          c1      = (float*)         carve((size_t)BATCH * HDIM * 4);
  float*          c2      = (float*)         carve((size_t)BATCH * HDIM * 4);
  float*          h0f     = (float*)         carve((size_t)BATCH * HDIM * 4);

  auto nblk = [](int n) { return (n + 255) / 256; };

  // --- pack weights to bf16 (N x K row-major for the GEMM B operand) ---
  pack_slice<<<nblk(EDIM * 2 * HDIM), 256, 0, stream>>>(attn_w1, w1q, EDIM, 2 * HDIM + EDIM, 0, 2 * HDIM);
  pack_slice<<<nblk(EDIM * EDIM), 256, 0, stream>>>(attn_w1, w1e, EDIM, 2 * HDIM + EDIM, 2 * HDIM, EDIM);
  pack_slice<<<nblk(4 * HDIM * HDIM), 256, 0, stream>>>(lstm_whh, whh0, 4 * HDIM, HDIM, 0, HDIM);
  pack2<<<nblk(4 * HDIM * 2 * HDIM), 256, 0, stream>>>(
      lstm_wih_rest, lstm_whh + (size_t)1 * 4 * HDIM * HDIM, W1, 4 * HDIM, HDIM, HDIM);
  pack2<<<nblk(4 * HDIM * 2 * HDIM), 256, 0, stream>>>(
      lstm_wih_rest + (size_t)4 * HDIM * HDIM, lstm_whh + (size_t)2 * 4 * HDIM * HDIM, W2, 4 * HDIM, HDIM, HDIM);

  // --- zero recurrent state (must happen every call: graph replays) ---
  zero_u32<<<nblk(BATCH * 2 * HDIM / 2), 256, 0, stream>>>((unsigned int*)qbf,  BATCH * 2 * HDIM / 2);
  zero_u32<<<nblk(BATCH * HDIM / 2),     256, 0, stream>>>((unsigned int*)h0bf, BATCH * HDIM / 2);
  zero_u32<<<nblk(BATCH * 2 * HDIM / 2), 256, 0, stream>>>((unsigned int*)xh1,  BATCH * 2 * HDIM / 2);
  zero_u32<<<nblk(BATCH * 2 * HDIM / 2), 256, 0, stream>>>((unsigned int*)xh2,  BATCH * 2 * HDIM / 2);
  zero_u32<<<nblk(BATCH * HDIM),         256, 0, stream>>>((unsigned int*)c0,   BATCH * HDIM);
  zero_u32<<<nblk(BATCH * HDIM),         256, 0, stream>>>((unsigned int*)c1,   BATCH * HDIM);
  zero_u32<<<nblk(BATCH * HDIM),         256, 0, stream>>>((unsigned int*)c2,   BATCH * HDIM);

  // --- enc_proj = input_encoded @ w1_e^T + attn_b1 (65024 x 256 x 256, bf16 WMMA) ---
  {
    dim3 grid(EDIM / 64, (BATCH * T1) / 64);
    gemm_wmma_bf16<true><<<grid, 128, 0, stream>>>(
        encproj, (const void*)input_encoded, w1e, attn_b1, BATCH * T1, EDIM, EDIM);
  }

  // --- 127 recurrent steps ---
  for (int t = 0; t < T1; ++t) {
    {   // qproj = [h0,c0] @ w1_q^T   (512 x 256 x 512)
      dim3 grid(EDIM / 64, BATCH / 64);
      gemm_wmma_bf16<false><<<grid, 128, 0, stream>>>(
          qproj, (const void*)qbf, w1q, nullptr, BATCH, EDIM, 2 * HDIM);
    }
    attn_kernel<<<BATCH, 256, 0, stream>>>(
        qproj, encproj, input_encoded, attn_w2, attn_b2, fc_w, fc_b, y_history, t, ctx, ytl);
    {   // L0 recurrent gates: h0 @ whh0^T   (512 x 1024 x 256)
      dim3 grid(4 * HDIM / 64, BATCH / 64);
      gemm_wmma_bf16<false><<<grid, 128, 0, stream>>>(
          g, (const void*)h0bf, whh0, nullptr, BATCH, 4 * HDIM, HDIM);
    }
    lstm_pw0<<<nblk(BATCH * HDIM), 256, 0, stream>>>(
        g, ytl, lstm_wih0, lstm_bih, lstm_bhh, c0, h0f, h0bf, qbf, xh1);
    {   // L1 gates: [h0,h1] @ [Wih1|Whh1]^T  (512 x 1024 x 512)
      dim3 grid(4 * HDIM / 64, BATCH / 64);
      gemm_wmma_bf16<false><<<grid, 128, 0, stream>>>(
          g, (const void*)xh1, W1, nullptr, BATCH, 4 * HDIM, 2 * HDIM);
    }
    lstm_pw12<<<nblk(BATCH * HDIM), 256, 0, stream>>>(
        g, lstm_bih + 4 * HDIM, lstm_bhh + 4 * HDIM, c1, xh1, xh2);
    {   // L2 gates: [h1,h2] @ [Wih2|Whh2]^T  (512 x 1024 x 512)
      dim3 grid(4 * HDIM / 64, BATCH / 64);
      gemm_wmma_bf16<false><<<grid, 128, 0, stream>>>(
          g, (const void*)xh2, W2, nullptr, BATCH, 4 * HDIM, 2 * HDIM);
    }
    lstm_pw12<<<nblk(BATCH * HDIM), 256, 0, stream>>>(
        g, lstm_bih + 8 * HDIM, lstm_bhh + 8 * HDIM, c2, xh2, nullptr);
  }

  final_kernel<<<BATCH, 256, 0, stream>>>(h0f, ctx, fcf_w, fcf_b, (float*)d_out);
}
